// AffinityOutputLayerPlus_1116691497782
// MI455X (gfx1250) — matrix operand order repaired and checked by reading, not compiled, not run on hardware
//
#include <hip/hip_runtime.h>
#include <hip/hip_bf16.h>
#include <stdint.h>

// ---------------------------------------------------------------------------
// Problem constants (from reference): B=256, LP=1024, LC=128, D=256
// ---------------------------------------------------------------------------
#define B_DIM   256
#define LP_DIM  1024
#define LC_DIM  128
#define D_DIM   256

#define TILE_M   32         // rows of X per workgroup
#define KSTAGE   64         // K elements staged per LDS slice (32 bf16 pairs)
#define KPAIRS   32         // dwords (bf16 pairs) per row per stage
#define LDP      36         // padded pair stride: 144B row pitch; 36 % 64 banks
                            // = 4-bank step -> conflict-free B128 fragments

#define BSTG     16         // float4 loads per thread per stage (B slice)
#define ASTG     2          // float4 loads per thread per stage (A slice)

typedef __attribute__((ext_vector_type(16))) __bf16 v16bf;
typedef __attribute__((ext_vector_type(8)))  float  v8f;

// One hardware instruction: packs two f32 into one dword of bf16 (RNE).
// Avoids the compiler's half-register merge (v_mov_b16/and/shl/or) pattern.
__device__ __forceinline__ uint32_t pack_bf16(float a, float b) {
    uint32_t r;
    asm("v_cvt_pk_bf16_f32 %0, %1, %2" : "=v"(r) : "v"(a), "v"(b));
    return r;
}

// ---------------------------------------------------------------------------
// Fused GEMM kernel:  f = relu(X @ W^T + bias)    (X: [B*L, D], W: [D, D])
//   MODE 0: per-row sum(f^2) -> sqrt -> outbuf[B*L]              (norms pass)
//   MODE 1: sum_l wvec[l] * f[l, d] -> atomicAdd outbuf[B*D]     (pool pass)
// f is never materialized (recompute strategy: tiny workspace, less HBM).
//
// Block = 256 threads = 8 waves. Each wave: 32 rows x 32 cols (2x2 register
// blocking -> 2 ds_load_b128 per WMMA, 4 independent WMMA chains). Block
// covers 32 rows x 256 cols; K=256 in 4 slices of 64, software-pipelined:
// slice s+1 global_load_b128 bursts are issued before the WMMA loop of
// slice s, hiding HBM/L2 latency under the matrix pipe.
// ---------------------------------------------------------------------------
template <int MODE>
__global__ __launch_bounds__(256)
void gemm_fused_kernel(const float* __restrict__ X,
                       const float* __restrict__ W,
                       const float* __restrict__ bias,
                       const float* __restrict__ wvec,   // MODE 1 only
                       float* __restrict__ outbuf,
                       int L)
{
    __shared__ __align__(16) uint32_t As[TILE_M][LDP];   // 4.5 KB
    __shared__ __align__(16) uint32_t Bs[D_DIM][LDP];    // 36 KB
    __shared__ float biasS[D_DIM];
    __shared__ float rowAcc[TILE_M];   // MODE 0: per-row sum of squares
    __shared__ float colAcc[D_DIM];    // MODE 1: per-column weighted sums
    __shared__ float wRow[TILE_M];     // MODE 1: softmax weights for our rows

    const int tid     = threadIdx.x;
    const int wg_row0 = blockIdx.x * TILE_M;       // global row base
    const int b       = wg_row0 / L;               // TILE_M | L -> no straddle

    if (tid < TILE_M) rowAcc[tid] = 0.0f;
    colAcc[tid] = 0.0f;
    biasS[tid]  = bias[tid];
    if (MODE == 1 && tid < TILE_M) wRow[tid] = wvec[(size_t)wg_row0 + tid];

    const int wave = tid >> 5;        // 0..7 -> column strip wave*32
    const int lane = tid & 31;
    const int lr   = lane & 15;
    const int hf   = lane >> 4;
    const int c0   = wave * 32;

    // Staging coordinates: 16 threads span one 256B row (fully coalesced).
    const int bcol = tid >> 4;        // 0..15 base col (stride 16 over k)
    const int bch  = tid & 15;        // float4 chunk within row

    v8f acc[2][2];                    // [mt][nt] : rows mt*16+, cols c0+nt*16+
    #pragma unroll
    for (int mt = 0; mt < 2; ++mt)
        #pragma unroll
        for (int nt = 0; nt < 2; ++nt)
            acc[mt][nt] = v8f{0.f,0.f,0.f,0.f,0.f,0.f,0.f,0.f};

    float4 bstg[BSTG];
    float4 astg[ASTG];

    // ---- prologue: load K-slice 0 into registers (burst of b128 loads) ----
    #pragma unroll
    for (int k = 0; k < BSTG; ++k)
        bstg[k] = *(const float4*)(W + (size_t)(bcol + k * 16) * D_DIM + bch * 4);
    #pragma unroll
    for (int k = 0; k < ASTG; ++k)
        astg[k] = *(const float4*)(X + (size_t)(wg_row0 + bcol + k * 16) * D_DIM + bch * 4);

    for (int s = 0; s < 4; ++s) {                 // K slices of 64
        __syncthreads();                          // LDS free (prev slice read)
        // ---- pack current slice registers into LDS (ds_store_b64) ----
        #pragma unroll
        for (int k = 0; k < BSTG; ++k) {
            uint2 u;
            u.x = pack_bf16(bstg[k].x, bstg[k].y);
            u.y = pack_bf16(bstg[k].z, bstg[k].w);
            *(uint2*)&Bs[bcol + k * 16][bch * 2] = u;
        }
        #pragma unroll
        for (int k = 0; k < ASTG; ++k) {
            uint2 u;
            u.x = pack_bf16(astg[k].x, astg[k].y);
            u.y = pack_bf16(astg[k].z, astg[k].w);
            *(uint2*)&As[bcol + k * 16][bch * 2] = u;
        }
        // ---- issue next slice's global loads (completion hidden by WMMAs) --
        if (s < 3) {
            const int ko = (s + 1) * KSTAGE;
            #pragma unroll
            for (int k = 0; k < BSTG; ++k)
                bstg[k] = *(const float4*)(W + (size_t)(bcol + k * 16) * D_DIM + ko + bch * 4);
            #pragma unroll
            for (int k = 0; k < ASTG; ++k)
                astg[k] = *(const float4*)(X + (size_t)(wg_row0 + bcol + k * 16) * D_DIM + ko + bch * 4);
        }
        __syncthreads();                          // LDS slice visible

        #pragma unroll
        for (int ks = 0; ks < 2; ++ks) {          // 2 x K=32 per slice
            // A fragments (ISA 7.12.2, 16-bit A 16x32):
            //   lane<16: K[0..7],K[16..23]; lane>=16: K[8..15],K[24..31]
            union { uint32_t u[8]; v16bf v; } fa[2], fb[2];
            const int pa = ks * 16 + hf * 4;
            #pragma unroll
            for (int mt = 0; mt < 2; ++mt) {
                const int arow = mt * 16 + lr;
                *(uint4*)&fa[mt].u[0] = *(const uint4*)&As[arow][pa];
                *(uint4*)&fa[mt].u[4] = *(const uint4*)&As[arow][pa + 8];
            }
            // B fragments (32x16): lane<16 -> col=lane K[0..7],K[16..23]; lane>=16 K[8..15],K[24..31]
            const int pb = ks * 16 + hf * 4;
            #pragma unroll
            for (int nt = 0; nt < 2; ++nt) {
                const int brow = c0 + nt * 16 + lr;
                *(uint4*)&fb[nt].u[0] = *(const uint4*)&Bs[brow][pb];
                *(uint4*)&fb[nt].u[4] = *(const uint4*)&Bs[brow][pb + 8];
            }
            // 4 WMMAs reuse each fragment twice -> 2 ds_load_b128 per WMMA
            #pragma unroll
            for (int mt = 0; mt < 2; ++mt)
                #pragma unroll
                for (int nt = 0; nt < 2; ++nt)
                    acc[mt][nt] = __builtin_amdgcn_wmma_f32_16x16x32_bf16(
                        false, fa[mt].v, false, fb[nt].v,
                        (short)0, acc[mt][nt], false, false);
        }
    }

    // Epilogue: C layout -> VGPR r holds (row = r + 8*hf, col = lane&15)
    if (MODE == 0) {
        #pragma unroll
        for (int mt = 0; mt < 2; ++mt) {
            #pragma unroll
            for (int nt = 0; nt < 2; ++nt) {
                const int col = c0 + nt * 16 + lr;
                #pragma unroll
                for (int r = 0; r < 8; ++r) {
                    const int row = mt * 16 + r + hf * 8;
                    float v = acc[mt][nt][r] + biasS[col];
                    v = v > 0.0f ? v : 0.0f;
                    atomicAdd(&rowAcc[row], v * v);          // ds_add_f32
                }
            }
        }
    } else {
        #pragma unroll
        for (int nt = 0; nt < 2; ++nt) {
            const int col = c0 + nt * 16 + lr;
            float colsum = 0.0f;
            #pragma unroll
            for (int mt = 0; mt < 2; ++mt) {
                #pragma unroll
                for (int r = 0; r < 8; ++r) {
                    const int row = mt * 16 + r + hf * 8;
                    float v = acc[mt][nt][r] + biasS[col];
                    v = v > 0.0f ? v : 0.0f;
                    colsum += v * wRow[row];
                }
            }
            atomicAdd(&colAcc[col], colsum);                 // ds_add_f32
        }
    }
    __syncthreads();

    if (MODE == 0) {
        if (tid < TILE_M) outbuf[(size_t)wg_row0 + tid] = sqrtf(rowAcc[tid]);
    } else {
        atomicAdd(&outbuf[(size_t)b * D_DIM + tid], colAcc[tid]);
    }
}

// ---------------------------------------------------------------------------
// Masked softmax over L (per batch).  Faithful to reference: max over ALL
// entries (masked included), exp zeroed where bool mask is true, +1e-6 in sum.
// ---------------------------------------------------------------------------
__global__ __launch_bounds__(256)
void softmax_kernel(const float* __restrict__ norms,
                    const unsigned char* __restrict__ bmask,  // jax bool = 1B
                    float* __restrict__ wout, int L)
{
    __shared__ float red[256];
    const int b = blockIdx.x;
    const int tid = threadIdx.x;
    const float* a = norms + (size_t)b * L;

    float mx = -3.4e38f;
    for (int i = tid; i < L; i += 256) mx = fmaxf(mx, a[i]);
    red[tid] = mx; __syncthreads();
    for (int s = 128; s > 0; s >>= 1) {
        if (tid < s) red[tid] = fmaxf(red[tid], red[tid + s]);
        __syncthreads();
    }
    mx = red[0]; __syncthreads();

    float sum = 0.0f;
    for (int i = tid; i < L; i += 256) {
        float e = expf(a[i] - mx);
        e = bmask[(size_t)b * L + i] ? 0.0f : e;
        wout[(size_t)b * L + i] = e;
        sum += e;
    }
    red[tid] = sum; __syncthreads();
    for (int s = 128; s > 0; s >>= 1) {
        if (tid < s) red[tid] += red[tid + s];
        __syncthreads();
    }
    const float inv = 1.0f / (red[0] + 1e-6f);
    for (int i = tid; i < L; i += 256) wout[(size_t)b * L + i] *= inv;
}

__global__ void zero_kernel(float* __restrict__ p, int n) {
    const int i = blockIdx.x * 256 + threadIdx.x;
    if (i < n) p[i] = 0.0f;
}

// ---------------------------------------------------------------------------
// out[b] = bf + sum_{c,p} relu(comp_sum[b,c] * prot_sum[b,p]) * Wf[c*D + p]
// ---------------------------------------------------------------------------
__global__ __launch_bounds__(256)
void final_kernel(const float* __restrict__ psum, const float* __restrict__ csum,
                  const float* __restrict__ Wf, const float* __restrict__ bf,
                  float* __restrict__ out)
{
    __shared__ float ps[D_DIM], cs[D_DIM], red[256];
    const int b = blockIdx.x;
    const int tid = threadIdx.x;
    ps[tid] = psum[(size_t)b * D_DIM + tid];
    cs[tid] = csum[(size_t)b * D_DIM + tid];
    __syncthreads();

    float acc = 0.0f;
    for (int idx = tid; idx < D_DIM * D_DIM; idx += 256) {
        const int ci = idx >> 8, pi = idx & 255;
        float v = cs[ci] * ps[pi];
        v = v > 0.0f ? v : 0.0f;
        acc += v * Wf[idx];
    }
    red[tid] = acc; __syncthreads();
    for (int s = 128; s > 0; s >>= 1) {
        if (tid < s) red[tid] += red[tid + s];
        __syncthreads();
    }
    if (tid == 0) out[b] = red[0] + bf[0];
}

// ---------------------------------------------------------------------------
// Launch: norms -> softmax -> zero -> weighted-sum GEMM -> final reduction
// Workspace use: ~2.8 MB of d_ws (norms + weights + pooled sums).
// ---------------------------------------------------------------------------
extern "C" void kernel_launch(void* const* d_in, const int* in_sizes, int n_in,
                              void* d_out, int out_size, void* d_ws, size_t ws_size,
                              hipStream_t stream) {
    const float*         prot  = (const float*)d_in[0];
    const float*         comp  = (const float*)d_in[1];
    const unsigned char* pmask = (const unsigned char*)d_in[2];
    const unsigned char* cmask = (const unsigned char*)d_in[3];
    const float*         Wp    = (const float*)d_in[4];
    const float*         bp    = (const float*)d_in[5];
    const float*         Wc    = (const float*)d_in[6];
    const float*         bc    = (const float*)d_in[7];
    const float*         Wf    = (const float*)d_in[8];
    const float*         bf    = (const float*)d_in[9];
    float* out = (float*)d_out;

    float* ws        = (float*)d_ws;
    float* prot_norm = ws;                                 // B*LP
    float* comp_norm = prot_norm + B_DIM * LP_DIM;         // B*LC
    float* prot_w    = comp_norm + B_DIM * LC_DIM;         // B*LP
    float* comp_w    = prot_w    + B_DIM * LP_DIM;         // B*LC
    float* prot_sum  = comp_w    + B_DIM * LC_DIM;         // B*D
    float* comp_sum  = prot_sum  + B_DIM * D_DIM;          // B*D

    // Pass 1: fused GEMM + row norms
    gemm_fused_kernel<0><<<B_DIM * LP_DIM / TILE_M, 256, 0, stream>>>(
        prot, Wp, bp, nullptr, prot_norm, LP_DIM);
    gemm_fused_kernel<0><<<B_DIM * LC_DIM / TILE_M, 256, 0, stream>>>(
        comp, Wc, bc, nullptr, comp_norm, LC_DIM);

    // Masked softmax over L
    softmax_kernel<<<B_DIM, 256, 0, stream>>>(prot_norm, pmask, prot_w, LP_DIM);
    softmax_kernel<<<B_DIM, 256, 0, stream>>>(comp_norm, cmask, comp_w, LC_DIM);

    // Zero pooled-sum accumulators (contiguous: prot_sum then comp_sum)
    zero_kernel<<<(2 * B_DIM * D_DIM + 255) / 256, 256, 0, stream>>>(
        prot_sum, 2 * B_DIM * D_DIM);

    // Pass 2: fused GEMM + softmax-weighted pooling
    gemm_fused_kernel<1><<<B_DIM * LP_DIM / TILE_M, 256, 0, stream>>>(
        prot, Wp, bp, prot_w, prot_sum, LP_DIM);
    gemm_fused_kernel<1><<<B_DIM * LC_DIM / TILE_M, 256, 0, stream>>>(
        comp, Wc, bc, comp_w, comp_sum, LC_DIM);

    // Final: relu(outer) . Wf + bf
    final_kernel<<<B_DIM, 256, 0, stream>>>(prot_sum, comp_sum, Wf, bf, out);
}
